// RegionProposalNetwork_76184129897269
// MI455X (gfx1250) — matrix operand order repaired and checked by reading, not compiled, not run on hardware
//
#include <hip/hip_runtime.h>
#include <hip/hip_bf16.h>

typedef _Float16 half_t;
typedef __attribute__((ext_vector_type(8)))  _Float16 v8h;
typedef __attribute__((ext_vector_type(16))) _Float16 v16h;
typedef __attribute__((ext_vector_type(8)))  float    v8f;

#define NBATCH   2
#define NPIX     13125      // pixels per batch (100^2 + 50^2 + 25^2)
#define NANCH    39375      // anchors per batch (3 * NPIX)
#define SORTN    65536
#define PRE_NMS_ 2000
#define POST_NMS_ 300

// ---- d_out layout (floats), concatenated return tuple ----
#define LOC_BASE 0          // 2*39375*4 = 315000
#define SC_BASE  315000     // 2*39375*2 = 157500
#define ROI_BASE 472500     // 600*4 = 2400
#define RI_BASE  474900     // 600
#define ANC_BASE 475500     // 39375*4 = 157500 -> total 633000

// ---- padded fp16 NHWC feature regions (offsets in halves) ----
// layout per level: [n][yp][xp][c], c = 256 contiguous halves
#define FP1 5326848         // after 2*102*102*256
#define FP2 6711296         // after + 2*52*52*256
#define FP_TOTAL 7084544    // + 2*27*27*256

// pixel / anchor offsets per level within one batch
#define P1 10000
#define P2 12500
#define A1 30000
#define A2 37500

#define TILE_TOTAL 1644     // 2 batches * (625 + 157 + 40) 16-pixel tiles

// ---------------------------------------------------------------------------
// K0: pack conv weights (256,256,3,3) f32 -> per-lane WMMA B fragments f16.
// Layout: wB[ntile(16)][chunk(72)][lane(32)][e(16)] halves.
// K ordering: r = tap*256 + c, tap = dy*3+dx. chunk covers 32 consecutive r.
// B 32x16 f16 layout: N=lane&15, hi=lane>>4; K = 8*hi+e (e<8), 16+8*hi+(e-8) (e>=8).
// ---------------------------------------------------------------------------
__global__ void pack_weights(const float* __restrict__ cw, half_t* __restrict__ wB) {
  int t = blockIdx.x * blockDim.x + threadIdx.x;
  if (t >= 589824) return;                      // 16*72*32*16
  int e     = t & 15;
  int lane  = (t >> 4) & 31;
  int chunk = (t >> 9) % 72;
  int nt    = t / 36864;                        // 72*32*16
  int N  = nt * 16 + (lane & 15);
  int kl = (e < 8) ? ((lane >> 4) * 8 + e) : (16 + (lane >> 4) * 8 + (e - 8));
  int K  = chunk * 32 + kl;
  int tap = K >> 8, c = K & 255;
  int dy = tap / 3, dx = tap - dy * 3;
  wB[t] = (half_t)cw[((N * 256 + c) * 3 + dy) * 3 + dx];
}

// ---------------------------------------------------------------------------
// K1: features f32 NCHW -> zero-padded f16 NHWC [n][H+2][W+2][256] per level.
// Consecutive threads write consecutive channels -> coalesced stores.
// ---------------------------------------------------------------------------
__global__ void pad_feats(const float* __restrict__ f0, const float* __restrict__ f1,
                          const float* __restrict__ f2, half_t* __restrict__ fp) {
  long long t = (long long)blockIdx.x * blockDim.x + threadIdx.x;
  if (t >= (long long)FP_TOTAL) return;
  int Hp, W; const float* src; long long loc;
  if (t < (long long)FP1)      { loc = t;       Hp = 102; W = 100; src = f0; }
  else if (t < (long long)FP2) { loc = t - FP1; Hp = 52;  W = 50;  src = f1; }
  else                         { loc = t - FP2; Hp = 27;  W = 25;  src = f2; }
  int c    = (int)(loc & 255);
  long long rest = loc >> 8;                    // n*Hp*Wp + yp*Wp + xp
  int PS   = Hp * Hp;
  int n    = (int)(rest / PS);
  int pos  = (int)(rest - (long long)n * PS);
  int yp = pos / Hp, xp = pos - yp * Hp;
  half_t v = (half_t)0.f;
  if (yp >= 1 && yp <= W && xp >= 1 && xp <= W)
    v = (half_t)src[(((long long)n * 256 + c) * W + (yp - 1)) * W + (xp - 1)];
  fp[t] = v;
}

// ---------------------------------------------------------------------------
// K2: implicit-GEMM 3x3 conv + bias + ReLU via v_wmma_f32_16x16x32_f16.
// One wave = 16 output pixels x 256 output channels (16 f32 accumulators).
// NHWC padded input => each lane's A fragment = two contiguous b128 loads.
// h output: f16, [n][pixel_concat][256] (channel-contiguous for the heads).
// ---------------------------------------------------------------------------
__global__ __launch_bounds__(256) void conv_wmma(
    const half_t* __restrict__ fpad, const v16h* __restrict__ wB,
    const float* __restrict__ conv_b, half_t* __restrict__ h) {
  int wave = threadIdx.x >> 5;
  int lane = threadIdx.x & 31;
  int tile = blockIdx.x * 8 + wave;
  if (tile >= TILE_TOTAL) return;
  int batch = tile / 822;
  int rt = tile - batch * 822;
  int level, tIn;
  if (rt < 625)      { level = 0; tIn = rt;       }
  else if (rt < 782) { level = 1; tIn = rt - 625; }
  else               { level = 2; tIn = rt - 782; }
  int W, fpoff, poff;
  if (level == 0)      { W = 100; fpoff = 0;   poff = 0;  }
  else if (level == 1) { W = 50;  fpoff = FP1; poff = P1; }
  else                 { W = 25;  fpoff = FP2; poff = P2; }
  int Wp = W + 2;
  int HW = W * W;
  int pixBase = tIn * 16;
  int mRow = lane & 15;
  int hi   = lane >> 4;                       // 0 or 1 (K-half select)
  int p = pixBase + mRow; if (p >= HW) p = HW - 1;   // clamp tail (store-guarded)
  int y = p / W, x = p - y * W;
  // NHWC: addr(tap, c) = fpoff + (((batch*Hp + y+dy)*Wp + x+dx)*256) + c
  size_t pixAddr = (size_t)fpoff
                 + ((((size_t)batch * Wp + y) * Wp + x) * 256)
                 + (size_t)(hi * 8);

  v8f acc[16];
#pragma unroll
  for (int t = 0; t < 16; ++t) acc[t] = (v8f){0,0,0,0,0,0,0,0};

  for (int chunk = 0; chunk < 72; ++chunk) {
    int tap = chunk >> 3, cb = chunk & 7;
    int dy = tap / 3, dx = tap - dy * 3;
    int c0 = cb * 32;
    // two contiguous 16-byte runs per lane: c = c0+hi*8+[0..7] and +16
    const v8h* pa = (const v8h*)(fpad + pixAddr
                                 + ((size_t)dy * Wp + dx) * 256 + c0);
    v8h alo = pa[0];
    v8h ahi = pa[2];
    v16h a = __builtin_shufflevector(alo, ahi,
                                     0, 1, 2, 3, 4, 5, 6, 7,
                                     8, 9, 10, 11, 12, 13, 14, 15);
    const v16h* wbc = wB + (size_t)chunk * 32 + lane;
#pragma unroll
    for (int t = 0; t < 16; ++t) {
      v16h b = wbc[(size_t)t * 72 * 32];
      acc[t] = __builtin_amdgcn_wmma_f32_16x16x32_f16(
          false, a, false, b, (short)0, acc[t], false, false);
    }
  }

  int M0 = hi * 8;                 // C/D f32 layout: lanes 16-31 hold M=8..15
  int Ncol = lane & 15;
#pragma unroll
  for (int t = 0; t < 16; ++t) {
    int N = t * 16 + Ncol;
    float bias = conv_b[N];
#pragma unroll
    for (int r = 0; r < 8; ++r) {
      int pix = pixBase + M0 + r;
      if (pix < HW) {
        float v = acc[t][r] + bias;
        if (v < 0.f) v = 0.f;
        h[((size_t)(batch * NPIX + poff + pix)) * 256 + N] = (half_t)v;
      }
    }
  }
}

// ---------------------------------------------------------------------------
// K3: 1x1 heads. Wave per pixel: 18 dot-products over 256 ch, butterfly
// reduce; write rpn_locs / rpn_scores into d_out, fg softmax into scratch.
// ---------------------------------------------------------------------------
__global__ __launch_bounds__(256) void head_proj(
    const half_t* __restrict__ h,
    const float* __restrict__ loc_w, const float* __restrict__ loc_b,
    const float* __restrict__ score_w, const float* __restrict__ score_b,
    float* __restrict__ out, float* __restrict__ fg) {
  int wave = threadIdx.x >> 5, lane = threadIdx.x & 31;
  int pix = blockIdx.x * 8 + wave;
  if (pix >= NBATCH * NPIX) return;
  float hv[8];
  const half_t* hp = h + (size_t)pix * 256 + lane * 8;
#pragma unroll
  for (int j = 0; j < 8; ++j) hv[j] = (float)hp[j];
  float res[18];
#pragma unroll
  for (int o = 0; o < 18; ++o) {
    const float* wr = (o < 12) ? (loc_w + o * 256) : (score_w + (o - 12) * 256);
    float s = 0.f;
#pragma unroll
    for (int j = 0; j < 8; ++j) s += hv[j] * wr[lane * 8 + j];
    s += __shfl_xor(s, 16); s += __shfl_xor(s, 8); s += __shfl_xor(s, 4);
    s += __shfl_xor(s, 2);  s += __shfl_xor(s, 1);
    res[o] = s + ((o < 12) ? loc_b[o] : score_b[o - 12]);
  }
  if (lane != 0) return;
  int batch = pix / NPIX, pp = pix - batch * NPIX;
  int p, aoff;
  if (pp < P1)      { p = pp;      aoff = 0;  }
  else if (pp < P2) { p = pp - P1; aoff = A1; }
  else              { p = pp - P2; aoff = A2; }
  size_t abase = (size_t)batch * NANCH + aoff + (size_t)p * 3;
#pragma unroll
  for (int o = 0; o < 12; ++o)
    out[LOC_BASE + (abase + o / 4) * 4 + (o & 3)] = res[o];
#pragma unroll
  for (int o = 0; o < 6; ++o)
    out[SC_BASE + (abase + o / 2) * 2 + (o & 1)] = res[12 + o];
#pragma unroll
  for (int ar = 0; ar < 3; ++ar) {
    float s0 = res[12 + ar * 2], s1 = res[13 + ar * 2];
    float m = fmaxf(s0, s1);
    float e0 = __expf(s0 - m), e1 = __expf(s1 - m);
    fg[abase + ar] = e1 / (e0 + e1);
  }
}

// ---------------------------------------------------------------------------
// K4: loc2bbox + clip + min-size mask; emit sortable keys (desc score, idx tie)
// ---------------------------------------------------------------------------
__global__ void decode_boxes(const float* __restrict__ anchors,
                             const float* __restrict__ out,
                             const float* __restrict__ fg,
                             float* __restrict__ roibuf,
                             unsigned long long* __restrict__ keys,
                             const int* __restrict__ pH, const int* __restrict__ pW) {
  int t = blockIdx.x * blockDim.x + threadIdx.x;
  if (t >= NBATCH * SORTN) return;
  int batch = t >> 16;
  int a = t & (SORTN - 1);
  unsigned long long* K = keys + (size_t)batch * SORTN;
  if (a >= NANCH) { K[a] = (0xFF800000ULL << 32) | (unsigned)a; return; }
  float Hf = (float)pH[0], Wf = (float)pW[0];
  float ay0 = anchors[a*4+0], ax0 = anchors[a*4+1];
  float ay1 = anchors[a*4+2], ax1 = anchors[a*4+3];
  float sh = ay1 - ay0, sw = ax1 - ax0;
  float cy = ay0 + 0.5f * sh, cx = ax0 + 0.5f * sw;
  const float* L = out + LOC_BASE + ((size_t)batch * NANCH + a) * 4;
  float dy = L[0], dx = L[1], dh = L[2], dw = L[3];
  float ncy = dy * sh + cy, ncx = dx * sw + cx;
  float nh = __expf(dh) * sh, nw = __expf(dw) * sw;
  float y0 = fminf(fmaxf(ncy - 0.5f * nh, 0.f), Hf);
  float x0 = fminf(fmaxf(ncx - 0.5f * nw, 0.f), Wf);
  float y1 = fminf(fmaxf(ncy + 0.5f * nh, 0.f), Hf);
  float x1 = fminf(fmaxf(ncx + 0.5f * nw, 0.f), Wf);
  float* R = roibuf + ((size_t)batch * NANCH + a) * 4;
  R[0] = y0; R[1] = x0; R[2] = y1; R[3] = x1;
  bool valid = ((y1 - y0) >= 16.f) && ((x1 - x0) >= 16.f);
  float sc = valid ? fg[(size_t)batch * NANCH + a] : -INFINITY;
  unsigned u = __float_as_uint(sc);
  unsigned enc = (u & 0x80000000u) ? u : ~(u | 0x80000000u);  // descending order
  K[a] = ((unsigned long long)enc << 32) | (unsigned)a;
}

// ---------------------------------------------------------------------------
// K5: one bitonic compare-exchange pass (ascending => descending scores)
// ---------------------------------------------------------------------------
__global__ void bitonic_step(unsigned long long* __restrict__ keys, int j, int k) {
  int t = blockIdx.x * blockDim.x + threadIdx.x;
  if (t >= NBATCH * SORTN) return;
  int b = t >> 16, i = t & (SORTN - 1);
  int ixj = i ^ j;
  if (ixj > i) {
    unsigned long long* K = keys + (size_t)b * SORTN;
    unsigned long long a = K[i], c = K[ixj];
    bool up = ((i & k) == 0);
    if ((a > c) == up) { K[i] = c; K[ixj] = a; }
  }
}

// ---------------------------------------------------------------------------
// K6: greedy NMS (matches the reference fori_loop) + ordered compaction.
// One block per batch; 2000 boxes + flags in LDS (~40 KB).
// ---------------------------------------------------------------------------
__global__ __launch_bounds__(256) void nms_kernel(
    const unsigned long long* __restrict__ keys,
    const float* __restrict__ roibuf, float* __restrict__ out) {
  __shared__ float bx[PRE_NMS_ * 4];
  __shared__ int  sup[PRE_NMS_];
  int batch = blockIdx.x;
  const unsigned long long* K = keys + (size_t)batch * SORTN;
  for (int i = threadIdx.x; i < PRE_NMS_; i += 256) {
    unsigned long long k = K[i];
    unsigned a = (unsigned)(k & 0xFFFFFFFFu);
    unsigned d = (unsigned)(k >> 32);
    const float* r = roibuf + ((size_t)batch * NANCH + a) * 4;
    bx[i*4+0] = r[0]; bx[i*4+1] = r[1]; bx[i*4+2] = r[2]; bx[i*4+3] = r[3];
    sup[i] = (d == 0xFF800000u) ? 1 : 0;     // -inf score => invalid
  }
  __syncthreads();
  for (int i = 0; i < PRE_NMS_; ++i) {
    if (!sup[i]) {
      float y0 = bx[i*4], x0 = bx[i*4+1], y1 = bx[i*4+2], x1 = bx[i*4+3];
      float ai = (y1 - y0) * (x1 - x0);
      for (int j = i + 1 + threadIdx.x; j < PRE_NMS_; j += 256) {
        float ty = fmaxf(y0, bx[j*4]),   tx = fmaxf(x0, bx[j*4+1]);
        float by = fminf(y1, bx[j*4+2]), bxr = fminf(x1, bx[j*4+3]);
        float ih = fmaxf(by - ty, 0.f),  iw = fmaxf(bxr - tx, 0.f);
        float inter = ih * iw;
        float aj = (bx[j*4+2] - bx[j*4]) * (bx[j*4+3] - bx[j*4+1]);
        float iou = inter / (ai + aj - inter + 1e-12f);
        if (iou > 0.7f) sup[j] = 1;
      }
    }
    __syncthreads();
  }
  for (int r = threadIdx.x; r < POST_NMS_; r += 256)
    out[RI_BASE + batch * POST_NMS_ + r] = (float)batch;
  if (threadIdx.x == 0) {
    float* R = out + ROI_BASE + (size_t)batch * POST_NMS_ * 4;
    int rank = 0;
    for (int i = 0; i < PRE_NMS_ && rank < POST_NMS_; ++i) {
      if (!sup[i]) {
        R[rank*4+0] = bx[i*4+0]; R[rank*4+1] = bx[i*4+1];
        R[rank*4+2] = bx[i*4+2]; R[rank*4+3] = bx[i*4+3];
        ++rank;
      }
    }
    for (; rank < POST_NMS_; ++rank) {
      R[rank*4+0] = 0.f; R[rank*4+1] = 0.f; R[rank*4+2] = 0.f; R[rank*4+3] = 0.f;
    }
  }
}

// ---------------------------------------------------------------------------
// K7: copy anchors into the output tail
// ---------------------------------------------------------------------------
__global__ void copy_anchors(const float* __restrict__ anchors, float* __restrict__ out) {
  int t = blockIdx.x * blockDim.x + threadIdx.x;
  if (t < NANCH * 4) out[ANC_BASE + t] = anchors[t];
}

// ---------------------------------------------------------------------------
extern "C" void kernel_launch(void* const* d_in, const int* in_sizes, int n_in,
                              void* d_out, int out_size, void* d_ws, size_t ws_size,
                              hipStream_t stream) {
  (void)in_sizes; (void)n_in; (void)out_size; (void)ws_size;
  const float* f0      = (const float*)d_in[0];
  const float* f1      = (const float*)d_in[1];
  const float* f2      = (const float*)d_in[2];
  const float* conv_w  = (const float*)d_in[3];
  const float* conv_b  = (const float*)d_in[4];
  const float* score_w = (const float*)d_in[5];
  const float* score_b = (const float*)d_in[6];
  const float* loc_w   = (const float*)d_in[7];
  const float* loc_b   = (const float*)d_in[8];
  const float* anchors = (const float*)d_in[9];
  const int*   img_h   = (const int*)d_in[10];
  const int*   img_w   = (const int*)d_in[11];
  float* out = (float*)d_out;
  char*  ws  = (char*)d_ws;

  // workspace layout (bytes)
  half_t* fpad = (half_t*)(ws);                                  // 14,169,088
  half_t* wB   = (half_t*)(ws + 14169088);                       //  1,179,648
  half_t* hbuf = (half_t*)(ws + 15348736);                       // 13,440,000
  float*  fg   = (float*) (ws + 28788736);                       //    315,000
  float*  roib = (float*) (ws + 29103736);                       //  1,260,000
  unsigned long long* keys = (unsigned long long*)(ws + 30363736); // 1,048,576

  pack_weights<<<2304, 256, 0, stream>>>(conv_w, wB);
  pad_feats<<<(FP_TOTAL + 255) / 256, 256, 0, stream>>>(f0, f1, f2, fpad);
  conv_wmma<<<(TILE_TOTAL + 7) / 8, 256, 0, stream>>>(fpad, (const v16h*)wB, conv_b, hbuf);
  head_proj<<<(NBATCH * NPIX + 7) / 8, 256, 0, stream>>>(hbuf, loc_w, loc_b,
                                                         score_w, score_b, out, fg);
  decode_boxes<<<(NBATCH * SORTN) / 256, 256, 0, stream>>>(anchors, out, fg, roib,
                                                           keys, img_h, img_w);
  for (int k = 2; k <= SORTN; k <<= 1)
    for (int j = k >> 1; j > 0; j >>= 1)
      bitonic_step<<<(NBATCH * SORTN) / 256, 256, 0, stream>>>(keys, j, k);
  nms_kernel<<<NBATCH, 256, 0, stream>>>(keys, roib, out);
  copy_anchors<<<(NANCH * 4 + 255) / 256, 256, 0, stream>>>(anchors, out);
}
